// ModelXLSTM_33998961116000
// MI455X (gfx1250) — matrix operand order repaired and checked by reading, not compiled, not run on hardware
//
#include <hip/hip_runtime.h>
#include <hip/hip_bf16.h>
#include <math.h>

// ---------------------------------------------------------------------------
// xLSTM forward for MI455X (gfx1250, wave32, WMMA).
// Dense math on v_wmma_f32_16x16x32_f16 (f16 in, f32 accumulate).
// LDS tiles are laid out so every WMMA fragment is built from aligned
// 128-bit ds loads (row stride = 40 halfs = 80B, a 16B multiple).
// ---------------------------------------------------------------------------

typedef __attribute__((ext_vector_type(16))) _Float16 v16h;
typedef __attribute__((ext_vector_type(8)))  _Float16 h8;
typedef __attribute__((ext_vector_type(8)))  float    v8f;

constexpr int kB    = 8;
constexpr int kS    = 1024;
constexpr int kD    = 256;
constexpr int kNH   = 4;
constexpr int kDHS  = 64;
constexpr int kDHM  = 128;
constexpr int kINNER = 512;
constexpr int kBS   = kB * kS;          // 8192 rows

__device__ __forceinline__ v8f wmma16(v16h a, v16h b, v8f c) {
  return __builtin_amdgcn_wmma_f32_16x16x32_f16(false, a, false, b, (short)0, c,
                                                false, false);
}
__device__ __forceinline__ v16h cat8(h8 lo, h8 hi) {
  return __builtin_shufflevector(lo, hi, 0, 1, 2, 3, 4, 5, 6, 7,
                                 8, 9, 10, 11, 12, 13, 14, 15);
}
__device__ __forceinline__ float sigmoidf_(float x) { return 1.f / (1.f + expf(-x)); }
__device__ __forceinline__ float siluf_(float x)    { return x * sigmoidf_(x); }
__device__ __forceinline__ float logsigf_(float x)  { return fminf(x, 0.f) - log1pf(expf(-fabsf(x))); }

// ---------------------------------------------------------------------------
// Generic GEMM: C[M,N] = A[M,K] * B[K,N] (+C), fp32 in/out, f16 WMMA compute.
// transB: B element (k,n) read from B[n*ldb + k] (for w(NH,out,in) layouts).
// Block: 256 threads (8 waves), 64x64 output tile, 32-K steps.
// LDS: A row-major, B stored transposed so both fragments are contiguous.
// ---------------------------------------------------------------------------
__launch_bounds__(256)
__global__ void gemm_f16wmma(const float* __restrict__ A, int lda,
                             const float* __restrict__ Bm, int ldb,
                             float* __restrict__ C, int ldc,
                             int M, int N, int K, int transB, int addC)
{
  __shared__ __align__(16) _Float16 As[64][40];    // [m][k]
  __shared__ __align__(16) _Float16 BsT[64][40];   // [n][k]
  const int tid  = threadIdx.x;
  const int wave = tid >> 5;
  const int lane = tid & 31;
  const int lm   = lane & 15;
  const int hl   = lane >> 4;
  const int m_off = (wave & 3) * 16;
  const int n_off = (wave >> 2) * 32;
  const int bm = blockIdx.x * 64;
  const int bn = blockIdx.y * 64;

  v8f acc0 = {}; v8f acc1 = {};

  for (int k0 = 0; k0 < K; k0 += 32) {
    { // A tile 64x32 -> As (vectorized 128-bit LDS store)
      const int r = tid >> 2, c = (tid & 3) * 8;
      const float* ap = A + (size_t)(bm + r) * lda + k0 + c;
      if (k0 + 32 < K) __builtin_prefetch(ap + 32, 0, 1);
      h8 hv;
#pragma unroll
      for (int j = 0; j < 8; ++j) hv[j] = (_Float16)ap[j];
      *(h8*)&As[r][c] = hv;
    }
    if (!transB) { // B tile 32x64, transpose into BsT
      const int r = tid >> 3, c = (tid & 7) * 8;
      const float* bp = Bm + (size_t)(k0 + r) * ldb + bn + c;
#pragma unroll
      for (int j = 0; j < 8; ++j) BsT[c + j][r] = (_Float16)bp[j];
    } else {       // already (n,k): straight vectorized copy
      const int n = tid >> 2, c = (tid & 3) * 8;
      const float* bp = Bm + (size_t)(bn + n) * ldb + k0 + c;
      h8 hv;
#pragma unroll
      for (int j = 0; j < 8; ++j) hv[j] = (_Float16)bp[j];
      *(h8*)&BsT[n][c] = hv;
    }
    __syncthreads();

    const int klo = hl * 16;   // A frag: K {klo..klo+15}
    const int kb  = hl * 16;   // B frag: K {kb..kb+15}
    const v16h a  = cat8(*(const h8*)&As[m_off + lm][klo],
                         *(const h8*)&As[m_off + lm][klo + 8]);
    const v16h b0 = cat8(*(const h8*)&BsT[n_off + lm][kb],
                         *(const h8*)&BsT[n_off + lm][kb + 8]);
    const v16h b1 = cat8(*(const h8*)&BsT[n_off + 16 + lm][kb],
                         *(const h8*)&BsT[n_off + 16 + lm][kb + 8]);
    acc0 = wmma16(a, b0, acc0);
    acc1 = wmma16(a, b1, acc1);
    __syncthreads();
  }

  const int row0 = bm + m_off + hl * 8;
#pragma unroll
  for (int r = 0; r < 8; ++r) {
    float* cp0 = C + (size_t)(row0 + r) * ldc + bn + n_off + lm;
    float* cp1 = cp0 + 16;
    float v0 = acc0[r], v1 = acc1[r];
    if (addC) { v0 += *cp0; v1 += *cp1; }
    *cp0 = v0; *cp1 = v1;
  }
}

// ---------------------------------------------------------------------------
// mLSTM parallel (flash-style with decay matrix), one wave per (b,h,16 rows).
// logD(s,t) = Lcum[s]-Lcum[t]+ig[t] (t<=s); running max/sum, WMMA for QK^T, PV.
// V tile (32x128) is staged transposed through LDS so P@V fragments are
// aligned 128-bit ds loads.
// ---------------------------------------------------------------------------
__launch_bounds__(32)
__global__ void attn_kernel(const float* __restrict__ q, const float* __restrict__ kk,
                            const float* __restrict__ vv, const float* __restrict__ ig,
                            const float* __restrict__ Lc, float* __restrict__ hout)
{
  __shared__ __align__(16) _Float16 Pl[16][40];    // P tile   [s][t]
  __shared__ __align__(16) _Float16 Vt[128][40];   // V tile^T [d][t]
  const int rt = blockIdx.x, hh = blockIdx.y, bb = blockIdx.z;
  const int lane = threadIdx.x & 31, lm = lane & 15, hl = lane >> 4;
  const int s0 = rt * 16;
  const size_t rowbase = ((size_t)bb * kS) * kINNER + (size_t)hh * kDHM;
  const int bhS = (bb * kNH + hh) * kS;
  const float qscale = 0.08838834764831845f; // 1/sqrt(128)
  const int klo = hl * 16;

  v16h qa[4];
  {
    const float* qrow = q + rowbase + (size_t)(s0 + lm) * kINNER;
#pragma unroll
    for (int c = 0; c < 4; ++c)
#pragma unroll
      for (int j = 0; j < 8; ++j) {
        qa[c][j]     = (_Float16)(qrow[c * 32 + klo + j] * qscale);
        qa[c][8 + j] = (_Float16)(qrow[c * 32 + klo + 8 + j] * qscale);
      }
  }

  float Lrow[8], rmax[8], rsum[8];
  v8f o[8];
#pragma unroll
  for (int r = 0; r < 8; ++r) {
    Lrow[r] = Lc[bhS + s0 + hl * 8 + r];
    rmax[r] = -1e30f; rsum[r] = 0.f;
  }
#pragma unroll
  for (int nt = 0; nt < 8; ++nt) o[nt] = (v8f){};

  for (int ct = 0; ct <= rt; ct += 2) {
    const int t0 = ct * 16, t1 = t0 + 16;

    // ---- stage V rows t0..t0+31 transposed into LDS (coalesced float4) ----
#pragma unroll 4
    for (int t = 0; t < 32; ++t) {
      const float4 v4 = *(const float4*)(vv + rowbase + (size_t)(t0 + t) * kINNER + lane * 4);
      Vt[lane * 4 + 0][t] = (_Float16)v4.x;
      Vt[lane * 4 + 1][t] = (_Float16)v4.y;
      Vt[lane * 4 + 2][t] = (_Float16)v4.z;
      Vt[lane * 4 + 3][t] = (_Float16)v4.w;
    }

    // ---- scores: two 16x16 tiles, K=128 contraction in 4 WMMA steps ----
    v8f sA = {}, sB = {};
    const float* k0row = kk + rowbase + (size_t)(t0 + lm) * kINNER;
    const float* k1row = kk + rowbase + (size_t)(t1 + lm) * kINNER;
#pragma unroll
    for (int c = 0; c < 4; ++c) {
      v16h b0, b1;
      const int kbase = c * 32 + hl * 16;
#pragma unroll
      for (int j = 0; j < 16; ++j) {
        b0[j] = (_Float16)k0row[kbase + j];
        b1[j] = (_Float16)k1row[kbase + j];
      }
      sA = wmma16(qa[c], b0, sA);
      sB = wmma16(qa[c], b1, sB);
    }

    // ---- decay + online rescale ----
    const int c0 = t0 + lm, c1 = t1 + lm;
    const float ig0 = ig[bhS + c0], ig1 = ig[bhS + c1];
    const float L0 = Lc[bhS + c0],  L1 = Lc[bhS + c1];
    float p0[8], p1[8], sc[8];
#pragma unroll
    for (int r = 0; r < 8; ++r) {
      const int srow = s0 + hl * 8 + r;
      float ld0 = (c0 <= srow) ? (Lrow[r] - L0 + ig0) : -1e30f;
      float ld1 = (c1 <= srow) ? (Lrow[r] - L1 + ig1) : -1e30f;
      float tm = fmaxf(ld0, ld1);
      tm = fmaxf(tm, __shfl_xor(tm, 8, 32));
      tm = fmaxf(tm, __shfl_xor(tm, 4, 32));
      tm = fmaxf(tm, __shfl_xor(tm, 2, 32));
      tm = fmaxf(tm, __shfl_xor(tm, 1, 32));
      const float nm = fmaxf(rmax[r], tm);
      const float s_ = expf(rmax[r] - nm);
      const float pv0 = sA[r] * expf(ld0 - nm);
      const float pv1 = sB[r] * expf(ld1 - nm);
      float ps = pv0 + pv1;
      ps += __shfl_xor(ps, 8, 32);
      ps += __shfl_xor(ps, 4, 32);
      ps += __shfl_xor(ps, 2, 32);
      ps += __shfl_xor(ps, 1, 32);
      rsum[r] = rsum[r] * s_ + ps;
      rmax[r] = nm;
      p0[r] = pv0; p1[r] = pv1; sc[r] = s_;
    }
#pragma unroll
    for (int nt = 0; nt < 8; ++nt)
#pragma unroll
      for (int r = 0; r < 8; ++r) o[nt][r] *= sc[r];

    // ---- stage P (16x32 f16) ----
#pragma unroll
    for (int r = 0; r < 8; ++r) {
      Pl[hl * 8 + r][lm]      = (_Float16)p0[r];
      Pl[hl * 8 + r][16 + lm] = (_Float16)p1[r];
    }
    __syncthreads();
    const v16h pa = cat8(*(const h8*)&Pl[lm][klo],
                         *(const h8*)&Pl[lm][klo + 8]);
    // ---- P @ V: 8 WMMAs across the 128-wide head dim ----
#pragma unroll
    for (int nt = 0; nt < 8; ++nt) {
      const v16h vb = cat8(*(const h8*)&Vt[nt * 16 + lm][hl * 16],
                           *(const h8*)&Vt[nt * 16 + lm][hl * 16 + 8]);
      o[nt] = wmma16(pa, vb, o[nt]);
    }
    __syncthreads();
  }

  float inv[8];
#pragma unroll
  for (int r = 0; r < 8; ++r) {
    const float nn = fmaxf(fabsf(rsum[r]), expf(-rmax[r]));
    inv[r] = 1.f / (nn + 1e-6f);
  }
#pragma unroll
  for (int nt = 0; nt < 8; ++nt)
#pragma unroll
    for (int r = 0; r < 8; ++r) {
      const int srow = s0 + hl * 8 + r;
      hout[rowbase + (size_t)srow * kINNER + nt * 16 + lm] = o[nt][r] * inv[r];
    }
}

// ---------------------------------------------------------------------------
// Elementwise / normalization / recurrence kernels
// ---------------------------------------------------------------------------
__global__ void ln_kernel(const float* __restrict__ x, const float* __restrict__ g,
                          float* __restrict__ y)
{
  __shared__ float red[kD];
  const int r = blockIdx.x, t = threadIdx.x;
  const float v = x[(size_t)r * kD + t];
  red[t] = v; __syncthreads();
  for (int st = 128; st > 0; st >>= 1) { if (t < st) red[t] += red[t + st]; __syncthreads(); }
  const float mu = red[0] * (1.f / kD); __syncthreads();
  const float d = v - mu;
  red[t] = d * d; __syncthreads();
  for (int st = 128; st > 0; st >>= 1) { if (t < st) red[t] += red[t + st]; __syncthreads(); }
  const float rs = rsqrtf(red[0] * (1.f / kD) + 1e-5f);
  y[(size_t)r * kD + t] = d * rs * g[t];
}

__global__ void conv_silu_kernel(const float* __restrict__ in, int ldin, int C,
                                 const float* __restrict__ w, const float* __restrict__ bias,
                                 float* __restrict__ out)
{
  const int id = blockIdx.x * blockDim.x + threadIdx.x;
  const int r = id / C, c = id - r * C;
  const int s = r & (kS - 1);
  float acc = bias[c];
#pragma unroll
  for (int k = 0; k < 4; ++k) {
    const int sp = s - 3 + k;
    if (sp >= 0) acc += in[(size_t)(r - 3 + k) * ldin + c] * w[c * 4 + k];
  }
  out[(size_t)r * C + c] = siluf_(acc);
}

__global__ void headwise_kernel(const float* __restrict__ in, int ldin,
                                const float* __restrict__ w, float* __restrict__ out)
{
  const int id = blockIdx.x * blockDim.x + threadIdx.x;   // kBS*kINNER threads
  const int r = id >> 9, o = id & 511;
  const int nb = o >> 2, oo = o & 3;
  const float* ir = in + (size_t)r * ldin + nb * 4;
  const float* wb = w + nb * 16 + oo * 4;
  out[(size_t)r * kINNER + o] = ir[0]*wb[0] + ir[1]*wb[1] + ir[2]*wb[2] + ir[3]*wb[3];
}

__global__ void igfg_kernel(const float* __restrict__ q, const float* __restrict__ k,
                            const float* __restrict__ v, const float* __restrict__ wi,
                            const float* __restrict__ bi, const float* __restrict__ wf,
                            const float* __restrict__ bf, float* __restrict__ ig,
                            float* __restrict__ fg)
{
  const int id = blockIdx.x * blockDim.x + threadIdx.x;   // kBS*kNH
  const int r = id >> 2, h = id & 3;
  float ai = bi[h], af = bf[h];
  const float* qr = q + (size_t)r * kINNER;
  const float* kr = k + (size_t)r * kINNER;
  const float* vr = v + (size_t)r * kINNER;
  for (int j = 0; j < kINNER; ++j) {
    ai += qr[j]*wi[j*kNH+h] + kr[j]*wi[(kINNER+j)*kNH+h] + vr[j]*wi[(2*kINNER+j)*kNH+h];
    af += qr[j]*wf[j*kNH+h] + kr[j]*wf[(kINNER+j)*kNH+h] + vr[j]*wf[(2*kINNER+j)*kNH+h];
  }
  const int b = r / kS, s = r - b * kS;
  ig[((size_t)b * kNH + h) * kS + s] = ai;
  fg[((size_t)b * kNH + h) * kS + s] = af;
}

__global__ void cumsum_kernel(const float* __restrict__ fg, float* __restrict__ Lc)
{
  if (threadIdx.x != 0) return;
  const int bh = blockIdx.x;
  const float* f = fg + (size_t)bh * kS;
  float* o = Lc + (size_t)bh * kS;
  float acc = 0.f;
  for (int s = 0; s < kS; ++s) { acc += logsigf_(f[s]); o[s] = acc; }
}

__global__ void mh_norm_gate_kernel(const float* __restrict__ hb, const float* __restrict__ up,
                                    const float* __restrict__ xc, const float* __restrict__ onorm,
                                    const float* __restrict__ skip, float* __restrict__ hmod)
{
  __shared__ float red[kDHM];
  const int r = blockIdx.x, hd = blockIdx.y, t = threadIdx.x;
  const float v = hb[(size_t)r * kINNER + hd * kDHM + t];
  red[t] = v; __syncthreads();
  for (int st = 64; st > 0; st >>= 1) { if (t < st) red[t] += red[t + st]; __syncthreads(); }
  const float mu = red[0] * (1.f / kDHM); __syncthreads();
  const float d = v - mu;
  red[t] = d * d; __syncthreads();
  for (int st = 64; st > 0; st >>= 1) { if (t < st) red[t] += red[t + st]; __syncthreads(); }
  const float rs = rsqrtf(red[0] * (1.f / kDHM) + 1e-5f);
  const int oi = hd * kDHM + t;
  float val = d * rs * onorm[oi] + skip[oi] * xc[(size_t)r * kINNER + oi];
  const float z = up[(size_t)r * 1024 + kINNER + oi];
  hmod[(size_t)r * kINNER + oi] = val * siluf_(z);
}

__launch_bounds__(64)
__global__ void slstm_scan_kernel(const float* __restrict__ gi, const float* __restrict__ gf,
                                  const float* __restrict__ gz, const float* __restrict__ go,
                                  const float* __restrict__ Ri, const float* __restrict__ Rf,
                                  const float* __restrict__ Rz, const float* __restrict__ Ro,
                                  const float* __restrict__ bi, const float* __restrict__ bf,
                                  const float* __restrict__ bz, const float* __restrict__ bo,
                                  float* __restrict__ hs)
{
  __shared__ float hsh[kDHS];
  const int hh = blockIdx.x, bb = blockIdx.y, e = threadIdx.x;
  const float* ri = Ri + hh * 4096; const float* rf = Rf + hh * 4096;
  const float* rz = Rz + hh * 4096; const float* ro = Ro + hh * 4096;
  const float bi_ = bi[hh*64+e], bf_ = bf[hh*64+e], bz_ = bz[hh*64+e], bo_ = bo[hh*64+e];
  float c = 0.f, n = 0.f, m = 0.f;
  hsh[e] = 0.f;
  __syncthreads();
  for (int s = 0; s < kS; ++s) {
    float reci = 0.f, recf = 0.f, recz = 0.f, reco = 0.f;
    for (int d = 0; d < kDHS; ++d) {
      const float hv = hsh[d];
      reci += hv * ri[d*64+e]; recf += hv * rf[d*64+e];
      recz += hv * rz[d*64+e]; reco += hv * ro[d*64+e];
    }
    const size_t gidx = ((size_t)bb * kS + s) * kD + hh * 64 + e;
    const float iraw = gi[gidx] + reci + bi_;
    const float fraw = gf[gidx] + recf + bf_;
    const float zraw = gz[gidx] + recz + bz_;
    const float oraw = go[gidx] + reco + bo_;
    const float lf = logsigf_(fraw) + m;
    const float mn = fmaxf(iraw, lf);
    const float ii = expf(iraw - mn);
    const float ff = expf(lf - mn);
    c = ff * c + ii * tanhf(zraw);
    n = ff * n + ii;
    m = mn;
    const float hn = sigmoidf_(oraw) * c / n;
    hs[gidx] = hn;
    __syncthreads();
    hsh[e] = hn;
    __syncthreads();
  }
}

__global__ void gn_residual_kernel(const float* __restrict__ hs, const float* __restrict__ gn,
                                   float* __restrict__ x)
{
  __shared__ float red[kDHS];
  const int r = blockIdx.x, hd = blockIdx.y, t = threadIdx.x;
  const float v = hs[(size_t)r * kD + hd * kDHS + t];
  red[t] = v; __syncthreads();
  for (int st = 32; st > 0; st >>= 1) { if (t < st) red[t] += red[t + st]; __syncthreads(); }
  const float mu = red[0] * (1.f / kDHS); __syncthreads();
  const float d = v - mu;
  red[t] = d * d; __syncthreads();
  for (int st = 32; st > 0; st >>= 1) { if (t < st) red[t] += red[t + st]; __syncthreads(); }
  const float rs = rsqrtf(red[0] * (1.f / kDHS) + 1e-5f);
  x[(size_t)r * kD + hd * kDHS + t] += d * rs * gn[hd * kDHS + t];
}

__global__ void ffact_kernel(const float* __restrict__ up, float* __restrict__ out)
{
  const int id = blockIdx.x * blockDim.x + threadIdx.x;   // kBS*kINNER
  const int r = id >> 9, j = id & 511;
  const float a = up[(size_t)r * 1024 + j];
  const float bg = up[(size_t)r * 1024 + kINNER + j];
  const float g = 0.5f * a * (1.f + erff(a * 0.70710678118654752f));
  out[(size_t)r * kINNER + j] = g * bg;
}

__global__ void embnorm_kernel(const float* __restrict__ seq_emb,
                               const float* __restrict__ tis_emb, float* __restrict__ scales)
{
  __shared__ float red[128];
  const int row = blockIdx.x, t = threadIdx.x;
  const float* p = (row < 65) ? (seq_emb + (size_t)row * 128) : (tis_emb + (size_t)(row - 65) * 128);
  const float v = p[t];
  red[t] = v * v; __syncthreads();
  for (int st = 64; st > 0; st >>= 1) { if (t < st) red[t] += red[t + st]; __syncthreads(); }
  if (t == 0) scales[row] = fminf(1.f, 5.0f / (sqrtf(red[0]) + 1e-7f));
}

__global__ void embed_kernel(const int* __restrict__ tok, const int* __restrict__ tissue,
                             const float* __restrict__ seq_emb, const float* __restrict__ tis_emb,
                             const float* __restrict__ scales, float* __restrict__ x)
{
  const int r = blockIdx.x, d = threadIdx.x;
  const int tk = tok[r];
  const int ti = tissue[r / kS];
  float val;
  if (d < 128) val = seq_emb[(size_t)tk * 128 + d] * scales[tk];
  else         val = tis_emb[(size_t)ti * 128 + (d - 128)] * scales[65 + ti];
  x[(size_t)r * kD + d] = (tk != 0) ? val : 0.f;
}

__global__ void head_kernel(const float* __restrict__ x, const int* __restrict__ seqlen,
                            const float* __restrict__ pn, const float* __restrict__ w1,
                            const float* __restrict__ b1, const float* __restrict__ w2,
                            const float* __restrict__ b2, float* __restrict__ out)
{
  __shared__ float row[kD];
  __shared__ float red[128];
  __shared__ float h1[128];
  const int bb = blockIdx.x, t = threadIdx.x;
  const int s = seqlen[bb] - 1;
  const float* xr = x + ((size_t)bb * kS + s) * kD;
  const float a0 = xr[t], a1 = xr[128 + t];
  red[t] = a0 + a1; __syncthreads();
  for (int st = 64; st > 0; st >>= 1) { if (t < st) red[t] += red[t + st]; __syncthreads(); }
  const float mu = red[0] / 256.f; __syncthreads();
  const float d0 = a0 - mu, d1 = a1 - mu;
  red[t] = d0 * d0 + d1 * d1; __syncthreads();
  for (int st = 64; st > 0; st >>= 1) { if (t < st) red[t] += red[t + st]; __syncthreads(); }
  const float rs = rsqrtf(red[0] / 256.f + 1e-5f); __syncthreads();
  row[t] = d0 * rs * pn[t];
  row[128 + t] = d1 * rs * pn[128 + t];
  __syncthreads();
  float acc = b1[t];
  for (int j = 0; j < kD; ++j) acc += row[j] * w1[j * 128 + t];
  h1[t] = fmaxf(acc, 0.f);
  __syncthreads();
  red[t] = h1[t] * w2[t]; __syncthreads();
  for (int st = 64; st > 0; st >>= 1) { if (t < st) red[t] += red[t + st]; __syncthreads(); }
  if (t == 0) out[bb] = red[0] + b2[0];
}

// ---------------------------------------------------------------------------
// Host orchestration
// ---------------------------------------------------------------------------
struct MBp { const float *ln,*w_up,*conv_w,*conv_b,*wq,*wk,*wv,*wi,*bi,*wf,*bf,*skip,*onorm,*w_down; };
struct SBp { const float *ln,*conv_w,*conv_b,*w_i,*w_f,*w_z,*w_o,*r_i,*r_f,*r_z,*r_o,
                         *b_i,*b_f,*b_z,*b_o,*gn,*ln2,*ff_up,*ff_down; };

static inline void gemm(hipStream_t st, const float* A, int lda, const float* Bm, int ldb,
                        float* C, int ldc, int M, int N, int K, int transB, int addC)
{
  dim3 g(M / 64, N / 64);
  gemm_f16wmma<<<g, 256, 0, st>>>(A, lda, Bm, ldb, C, ldc, M, N, K, transB, addC);
}

extern "C" void kernel_launch(void* const* d_in, const int* in_sizes, int n_in,
                              void* d_out, int out_size, void* d_ws, size_t ws_size,
                              hipStream_t stream)
{
  (void)in_sizes; (void)n_in; (void)out_size; (void)ws_size;
  int pi = 0;
  const int* tok       = (const int*)d_in[pi++];
  const int* tissue_id = (const int*)d_in[pi++];
  const int* seq_len   = (const int*)d_in[pi++];
  const float* tissue_emb = (const float*)d_in[pi++];
  const float* seq_emb    = (const float*)d_in[pi++];

  MBp mb[3]; SBp sb; int mi = 0;
  for (int blk = 0; blk < 4; ++blk) {
    if (blk == 1) {
      sb.ln=(const float*)d_in[pi++]; sb.conv_w=(const float*)d_in[pi++]; sb.conv_b=(const float*)d_in[pi++];
      sb.w_i=(const float*)d_in[pi++]; sb.w_f=(const float*)d_in[pi++]; sb.w_z=(const float*)d_in[pi++];
      sb.w_o=(const float*)d_in[pi++]; sb.r_i=(const float*)d_in[pi++]; sb.r_f=(const float*)d_in[pi++];
      sb.r_z=(const float*)d_in[pi++]; sb.r_o=(const float*)d_in[pi++]; sb.b_i=(const float*)d_in[pi++];
      sb.b_f=(const float*)d_in[pi++]; sb.b_z=(const float*)d_in[pi++]; sb.b_o=(const float*)d_in[pi++];
      sb.gn=(const float*)d_in[pi++]; sb.ln2=(const float*)d_in[pi++]; sb.ff_up=(const float*)d_in[pi++];
      sb.ff_down=(const float*)d_in[pi++];
    } else {
      MBp& m = mb[mi++];
      m.ln=(const float*)d_in[pi++]; m.w_up=(const float*)d_in[pi++]; m.conv_w=(const float*)d_in[pi++];
      m.conv_b=(const float*)d_in[pi++]; m.wq=(const float*)d_in[pi++]; m.wk=(const float*)d_in[pi++];
      m.wv=(const float*)d_in[pi++]; m.wi=(const float*)d_in[pi++]; m.bi=(const float*)d_in[pi++];
      m.wf=(const float*)d_in[pi++]; m.bf=(const float*)d_in[pi++]; m.skip=(const float*)d_in[pi++];
      m.onorm=(const float*)d_in[pi++]; m.w_down=(const float*)d_in[pi++];
    }
  }
  const float* post_norm = (const float*)d_in[pi++];
  const float* pred_w1   = (const float*)d_in[pi++];
  const float* pred_b1   = (const float*)d_in[pi++];
  const float* pred_w2   = (const float*)d_in[pi++];
  const float* pred_b2   = (const float*)d_in[pi++];

  // workspace layout (fp32)
  float* ws  = (float*)d_ws;
  float* x   = ws;                          // [8192,256]
  float* xln = x   + (size_t)kBS * kD;      // [8192,256]
  float* up  = xln + (size_t)kBS * kD;      // [8192,1024]
  float* xc  = up  + (size_t)kBS * 1024;    // [8192,512]
  float* qb  = xc  + (size_t)kBS * kINNER;  // [8192,512]
  float* kb  = qb  + (size_t)kBS * kINNER;  // [8192,512]
  float* vb  = kb  + (size_t)kBS * kINNER;  // [8192,512]
  float* hb  = vb  + (size_t)kBS * kINNER;  // [8192,512]
  float* igb = hb  + (size_t)kBS * kINNER;  // [B*NH*S]
  float* fgb = igb + (size_t)kB * kNH * kS;
  float* Lcb = fgb + (size_t)kB * kNH * kS;
  float* scl = Lcb + (size_t)kB * kNH * kS; // [95]

  embnorm_kernel<<<95, 128, 0, stream>>>(seq_emb, tissue_emb, scl);
  embed_kernel<<<kBS, 256, 0, stream>>>(tok, tissue_id, seq_emb, tissue_emb, scl, x);

  auto run_mlstm = [&](const MBp& p) {
    ln_kernel<<<kBS, 256, 0, stream>>>(x, p.ln, xln);
    gemm(stream, xln, kD, p.w_up, 1024, up, 1024, kBS, 1024, kD, 0, 0);
    conv_silu_kernel<<<(kBS * kINNER) / 256, 256, 0, stream>>>(up, 1024, kINNER, p.conv_w, p.conv_b, xc);
    headwise_kernel<<<(kBS * kINNER) / 256, 256, 0, stream>>>(xc, kINNER, p.wq, qb);
    headwise_kernel<<<(kBS * kINNER) / 256, 256, 0, stream>>>(xc, kINNER, p.wk, kb);
    headwise_kernel<<<(kBS * kINNER) / 256, 256, 0, stream>>>(up, 1024, p.wv, vb);
    igfg_kernel<<<(kBS * kNH) / 256, 256, 0, stream>>>(qb, kb, vb, p.wi, p.bi, p.wf, p.bf, igb, fgb);
    cumsum_kernel<<<kB * kNH, 32, 0, stream>>>(fgb, Lcb);
    attn_kernel<<<dim3(kS / 16, kNH, kB), 32, 0, stream>>>(qb, kb, vb, igb, Lcb, hb);
    mh_norm_gate_kernel<<<dim3(kBS, kNH), 128, 0, stream>>>(hb, up, xc, p.onorm, p.skip, kb);
    gemm(stream, kb, kINNER, p.w_down, kD, x, kD, kBS, kD, kINNER, 0, 1);
  };

  run_mlstm(mb[0]);

  // ---- sLSTM block ----
  ln_kernel<<<kBS, 256, 0, stream>>>(x, sb.ln, xln);
  conv_silu_kernel<<<(kBS * kD) / 256, 256, 0, stream>>>(xln, kD, kD, sb.conv_w, sb.conv_b, xc);
  for (int h = 0; h < kNH; ++h) {
    gemm(stream, xc  + h * 64, kD, sb.w_i + h * 4096, 64, qb + h * 64, kD, kBS, 64, 64, 1, 0);
    gemm(stream, xc  + h * 64, kD, sb.w_f + h * 4096, 64, kb + h * 64, kD, kBS, 64, 64, 1, 0);
    gemm(stream, xln + h * 64, kD, sb.w_z + h * 4096, 64, vb + h * 64, kD, kBS, 64, 64, 1, 0);
    gemm(stream, xln + h * 64, kD, sb.w_o + h * 4096, 64, hb + h * 64, kD, kBS, 64, 64, 1, 0);
  }
  slstm_scan_kernel<<<dim3(kNH, kB), 64, 0, stream>>>(qb, kb, vb, hb,
      sb.r_i, sb.r_f, sb.r_z, sb.r_o, sb.b_i, sb.b_f, sb.b_z, sb.b_o, up);
  gn_residual_kernel<<<dim3(kBS, kNH), 64, 0, stream>>>(up, sb.gn, x);
  // ---- FFN of the sLSTM block ----
  ln_kernel<<<kBS, 256, 0, stream>>>(x, sb.ln2, xln);
  gemm(stream, xln, kD, sb.ff_up, 1024, up, 1024, kBS, 1024, kD, 0, 0);
  ffact_kernel<<<(kBS * kINNER) / 256, 256, 0, stream>>>(up, xc);
  gemm(stream, xc, kINNER, sb.ff_down, kD, x, kD, kBS, kD, kINNER, 0, 1);

  run_mlstm(mb[1]);
  run_mlstm(mb[2]);

  head_kernel<<<kB, 128, 0, stream>>>(x, seq_len, post_norm, pred_w1, pred_b1,
                                      pred_w2, pred_b2, (float*)d_out);
}
